// SelfAttentionBlock_48790828482901
// MI455X (gfx1250) — matrix operand order repaired and checked by reading, not compiled, not run on hardware
//
#include <hip/hip_runtime.h>
#include <hip/hip_bf16.h>

// ---------------------------------------------------------------------------
// Types for CDNA5 WMMA (gfx1250, wave32)
// ---------------------------------------------------------------------------
typedef __attribute__((ext_vector_type(16))) __bf16 v16bf;
typedef __attribute__((ext_vector_type(8)))  float  v8f;
typedef int i4v __attribute__((vector_size(16)));

union Frag {
    v16bf v;
    unsigned int u[8];
    uint4 q[2];
};

__device__ __forceinline__ v8f vzero8() {
    v8f z;
#pragma unroll
    for (int i = 0; i < 8; ++i) z[i] = 0.0f;
    return z;
}

__device__ __forceinline__ v8f bf16_wmma(const Frag& a, const Frag& b, v8f c) {
    return __builtin_amdgcn_wmma_f32_16x16x32_bf16(
        /*neg_a=*/false, a.v, /*neg_b=*/false, b.v,
        /*c_mod=*/(short)0, c, /*reuse_a=*/false, /*reuse_b=*/false);
}

__device__ __forceinline__ void load_frag8(Frag& f, const unsigned int* p) {
    f.q[0] = *(const uint4*)p;        // ds_load_b128
    f.q[1] = *(const uint4*)(p + 4);  // ds_load_b128
}

// ---- CDNA5 async global->LDS copy (guarded; falls back to plain copy) ----
#if __has_builtin(__builtin_amdgcn_global_load_async_to_lds_b128) && \
    __has_builtin(__builtin_amdgcn_s_wait_asynccnt)
#define HAS_ASYNC 1
#else
#define HAS_ASYNC 0
#endif

__device__ __forceinline__ void async_copy16(void* lds_dst, const void* gsrc) {
#if HAS_ASYNC
    __builtin_amdgcn_global_load_async_to_lds_b128(
        (__attribute__((address_space(1))) i4v*)gsrc,
        (__attribute__((address_space(3))) i4v*)lds_dst, 0, 0);
#else
    *(uint4*)lds_dst = *(const uint4*)gsrc;
#endif
}
__device__ __forceinline__ void async_wait() {
#if HAS_ASYNC
    __builtin_amdgcn_s_wait_asynccnt(0);
#endif
}

__device__ __forceinline__ unsigned short f2bf(float f) {
    unsigned int u = __float_as_uint(f);
    unsigned int r = u + 0x7FFFu + ((u >> 16) & 1u);   // round-to-nearest-even
    return (unsigned short)(r >> 16);
}

__device__ __forceinline__ float gelu_tanh(float x) {
    const float c = 0.7978845608028654f;               // sqrt(2/pi)
    float t = c * (x + 0.044715f * x * x * x);
    return 0.5f * x * (1.0f + tanhf(t));
}

// ---------------------------------------------------------------------------
// Problem constants
// ---------------------------------------------------------------------------
constexpr int kTok = 16384;   // (4*16) * (16*16) tokens
constexpr int kNE  = 768;
constexpr int kQKV = 2304;
constexpr int kFF  = 3072;
constexpr int kHeads = 12;
constexpr int kHD  = 64;
constexpr int kT   = 256;

// Workspace layout (bytes)
constexpr size_t OFF_X1F  = 0;
constexpr size_t SZ_X1F   = (size_t)kTok * kNE * 4;
constexpr size_t OFF_X1H  = OFF_X1F + SZ_X1F;
constexpr size_t SZ_X1H   = (size_t)kTok * kNE * 2;
constexpr size_t OFF_QKVH = OFF_X1H + SZ_X1H;
constexpr size_t SZ_QKVH  = (size_t)kTok * kQKV * 2;
constexpr size_t OFF_YH   = OFF_QKVH + SZ_QKVH;
constexpr size_t SZ_YH    = (size_t)kTok * kNE * 2;
constexpr size_t OFF_HH   = OFF_QKVH;                 // reuses qkv+y region
constexpr size_t OFF_X2F  = OFF_YH + SZ_YH;
constexpr size_t SZ_X2F   = SZ_X1F;
constexpr size_t OFF_WQKV = OFF_X2F + SZ_X2F;
constexpr size_t SZ_WQKV  = (size_t)kNE * kQKV * 2;
constexpr size_t OFF_WP   = OFF_WQKV + SZ_WQKV;
constexpr size_t SZ_WP    = (size_t)kNE * kNE * 2;
constexpr size_t OFF_WFC  = OFF_WP + SZ_WP;
constexpr size_t SZ_WFC   = (size_t)kNE * kFF * 2;
constexpr size_t OFF_WM   = OFF_WFC + SZ_WFC;

// Attention LDS: KF 32KB + QF 8KB + VF 32KB + S 65KB + rsum
constexpr int ATTN_SMEM = (8192 + 2048 + 8192) * 4 + 64 * 260 * 4 + 64 * 4;  // 140544

// ---------------------------------------------------------------------------
__global__ void __launch_bounds__(256) cvt_bf16_kernel(const float* __restrict__ src,
                                                       unsigned short* __restrict__ dst, int n) {
    int i = blockIdx.x * 256 + threadIdx.x;
    if (i < n) dst[i] = f2bf(src[i]);
}

// ---------------------------------------------------------------------------
__device__ __forceinline__ float block_sum256(float s, float* red) {
#pragma unroll
    for (int o = 16; o > 0; o >>= 1) s += __shfl_down(s, o, 32);
    int wave = threadIdx.x >> 5, lane = threadIdx.x & 31;
    if (lane == 0) red[wave] = s;
    __syncthreads();
    float t = (threadIdx.x < 8) ? red[threadIdx.x] : 0.0f;
    if (threadIdx.x < 32) {
#pragma unroll
        for (int o = 4; o > 0; o >>= 1) t += __shfl_down(t, o, 32);
    }
    if (threadIdx.x == 0) red[0] = t;
    __syncthreads();
    float r = red[0];
    __syncthreads();
    return r;
}

// ---------------------------------------------------------------------------
// LN1: transpose [v,b,c,h,w] -> [tok,c] + layernorm; emit fp32 + bf16
// ---------------------------------------------------------------------------
__global__ void __launch_bounds__(256) ln1_kernel(const float* __restrict__ x,
                                                  const float* __restrict__ scale,
                                                  const float* __restrict__ bias,
                                                  float* __restrict__ x1f,
                                                  unsigned short* __restrict__ x1h) {
    __shared__ float red[8];
    const int t  = blockIdx.x;
    const int vb = t >> 8, hw = t & 255;
    const float* base = x + (size_t)vb * kNE * 256 + hw;
    const int tid = threadIdx.x;

    float v[3];
#pragma unroll
    for (int i = 0; i < 3; ++i) v[i] = base[(size_t)(tid + i * 256) * 256];

    float mu = block_sum256(v[0] + v[1] + v[2], red) * (1.0f / 768.0f);
    float d0 = v[0] - mu, d1 = v[1] - mu, d2 = v[2] - mu;
    float var = block_sum256(d0 * d0 + d1 * d1 + d2 * d2, red) * (1.0f / 768.0f);
    float rstd = rsqrtf(var + 1e-5f);

#pragma unroll
    for (int i = 0; i < 3; ++i) {
        int c = tid + i * 256;
        float y = (v[i] - mu) * rstd * scale[c] + bias[c];
        x1f[(size_t)t * kNE + c] = y;
        x1h[(size_t)t * kNE + c] = f2bf(y);
    }
}

// ---------------------------------------------------------------------------
__global__ void __launch_bounds__(256) ln2_kernel(const float* __restrict__ xin,
                                                  const float* __restrict__ scale,
                                                  const float* __restrict__ bias,
                                                  unsigned short* __restrict__ xout) {
    __shared__ float red[8];
    const int t = blockIdx.x;
    const int tid = threadIdx.x;
    const float* base = xin + (size_t)t * kNE;

    float v[3];
#pragma unroll
    for (int i = 0; i < 3; ++i) v[i] = base[tid + i * 256];

    float mu = block_sum256(v[0] + v[1] + v[2], red) * (1.0f / 768.0f);
    float d0 = v[0] - mu, d1 = v[1] - mu, d2 = v[2] - mu;
    float var = block_sum256(d0 * d0 + d1 * d1 + d2 * d2, red) * (1.0f / 768.0f);
    float rstd = rsqrtf(var + 1e-5f);

#pragma unroll
    for (int i = 0; i < 3; ++i) {
        int c = tid + i * 256;
        float y = (v[i] - mu) * rstd * scale[c] + bias[c];
        xout[(size_t)t * kNE + c] = f2bf(y);
    }
}

// ---------------------------------------------------------------------------
// bf16 WMMA GEMM, fragment-order LDS tiles.
//   A tile: Af[mtile 0..7][lane 0..31][8 dwords]   (async 16B copies)
//   B tile: Bf[ntile 0..7][lane 0..31][8 dwords]   (2-row packed dword scatter)
//   EPI 0: bf16  | 1: fp32+residual | 2: gelu->bf16 | 3: fp32+residual scattered
// ---------------------------------------------------------------------------
template <int EPI>
__global__ void __launch_bounds__(256) gemm_bf16_kernel(
    const unsigned short* __restrict__ A,   // [M,K] bf16
    const unsigned short* __restrict__ W,   // [K,N] bf16
    const float* __restrict__ bias,         // [N]
    const float* __restrict__ residual,     // [M,N] fp32 (EPI 1/3)
    float* __restrict__ outF,
    unsigned short* __restrict__ outH,
    int M, int N, int K) {
    __shared__ unsigned int Af[2048];   // 8KB
    __shared__ unsigned int Bf[2048];   // 8KB

    const int tid  = threadIdx.x;
    const int wave = tid >> 5, lane = tid & 31;
    const int grp  = lane >> 4, lr = lane & 15;
    const int wm   = wave & 1,  wn = wave >> 1;
    const int mBase = blockIdx.y * 128;
    const int nBase = blockIdx.x * 128;

    v8f acc[4][2];
#pragma unroll
    for (int i = 0; i < 4; ++i)
#pragma unroll
        for (int j = 0; j < 2; ++j) acc[i][j] = vzero8();

    const int aRow = tid >> 1, aSeg = tid & 1;          // A: row, 16-half segment
    const int aMt = aRow >> 4, aLr = aRow & 15;
    const int bPair = tid >> 4, bNg = tid & 15;         // B: k-pair, n-group-of-8
    const int bGrp = (bPair >> 2) & 1, bJ = (bPair & 3) + 4 * (bPair >> 3);
    const int bNt = bNg >> 1, bLr0 = (bNg & 1) * 8;

    for (int k0 = 0; k0 < K; k0 += 32) {
        // ---- A tile: two contiguous 16B chunks -> fragment slots (async) ----
        const unsigned short* ga = A + (size_t)(mBase + aRow) * K + k0 + aSeg * 16;
        async_copy16(&Af[(aMt * 32 + aLr) * 8 + aSeg * 4], ga);            // grp0
        async_copy16(&Af[(aMt * 32 + aLr + 16) * 8 + aSeg * 4], ga + 8);   // grp1

        // ---- B tile: read rows k,k+1 x 8 cols, pack dwords, scatter ----
        {
            const unsigned short* g0 = W + (size_t)(k0 + bPair * 2) * N + nBase + bNg * 8;
            const unsigned short* g1 = g0 + N;
            alignas(16) unsigned short h0[8], h1[8];
            *(uint4*)h0 = *(const uint4*)g0;
            *(uint4*)h1 = *(const uint4*)g1;
#pragma unroll
            for (int e = 0; e < 8; ++e) {
                unsigned int dw = (unsigned int)h0[e] | ((unsigned int)h1[e] << 16);
                Bf[(bNt * 32 + bLr0 + e + 16 * bGrp) * 8 + bJ] = dw;
            }
        }
        if (k0 + 32 < K) {
            __builtin_prefetch(A + (size_t)(mBase + aRow) * K + k0 + 32 + aSeg * 16, 0, 1);
            __builtin_prefetch(W + (size_t)(k0 + 32 + bPair * 2) * N + nBase + bNg * 8, 0, 1);
        }
        async_wait();
        __syncthreads();

        // ---- fragments: 2x ds_load_b128 each ----
        Frag af[4];
#pragma unroll
        for (int mi = 0; mi < 4; ++mi)
            load_frag8(af[mi], &Af[((wm * 4 + mi) * 32 + lane) * 8]);
        Frag bfr[2];
#pragma unroll
        for (int ni = 0; ni < 2; ++ni)
            load_frag8(bfr[ni], &Bf[((wn * 2 + ni) * 32 + lane) * 8]);
#pragma unroll
        for (int mi = 0; mi < 4; ++mi)
#pragma unroll
            for (int ni = 0; ni < 2; ++ni)
                acc[mi][ni] = bf16_wmma(af[mi], bfr[ni], acc[mi][ni]);
        __syncthreads();
    }

    // ---- epilogue (C/D layout: row = r + 8*grp, col = lane%16) ----
#pragma unroll
    for (int mi = 0; mi < 4; ++mi) {
#pragma unroll
        for (int ni = 0; ni < 2; ++ni) {
            int colg = nBase + wn * 32 + ni * 16 + lr;
            float bv = bias[colg];
#pragma unroll
            for (int r = 0; r < 8; ++r) {
                int rowg = mBase + wm * 64 + mi * 16 + r + 8 * grp;
                float val = acc[mi][ni][r] + bv;
                if (EPI == 0) {
                    outH[(size_t)rowg * N + colg] = f2bf(val);
                } else if (EPI == 1) {
                    outF[(size_t)rowg * N + colg] = residual[(size_t)rowg * N + colg] + val;
                } else if (EPI == 2) {
                    outH[(size_t)rowg * N + colg] = f2bf(gelu_tanh(val));
                } else {
                    float o = residual[(size_t)rowg * N + colg] + val;
                    int vb = rowg >> 8, hw = rowg & 255;
                    outF[((size_t)vb * kNE + colg) * 256 + hw] = o;
                }
            }
        }
    }
}

// ---------------------------------------------------------------------------
// Attention: workgroup per (head, batch). K/V fragment-packed in LDS once,
// queries in 4 chunks of 64 rows. S = QK^T and Y = PV via WMMA.
//   KF[ks 0..1][ntile 0..15][lane][8dw]  (B-operand order, async staged)
//   QF[ks 0..1][mtile 0..3 ][lane][8dw]  (A-operand order, async staged)
//   VF[ks 0..7][dtile 0..3 ][lane][8dw]  (B-operand order, packed scatter)
// ---------------------------------------------------------------------------
__global__ void __launch_bounds__(256) attn_kernel(const unsigned short* __restrict__ qkv,
                                                   unsigned short* __restrict__ y) {
    extern __shared__ char smem[];
    unsigned int* KF = (unsigned int*)smem;        // 8192 dwords
    unsigned int* QF = KF + 8192;                  // 2048 dwords
    unsigned int* VF = QF + 2048;                  // 8192 dwords
    float* Sf   = (float*)(VF + 8192);             // [64][260] fp32
    float* rsum = Sf + 64 * 260;                   // [64]

    const int tid  = threadIdx.x;
    const int wave = tid >> 5, lane = tid & 31;
    const int grp  = lane >> 4, lr = lane & 15;
    const int head = blockIdx.x;
    const int vb   = blockIdx.y;

    const unsigned short* qk = qkv + (size_t)(vb * kT) * kQKV;
    const int qoff = head * kHD;
    const int koff = kNE + head * kHD;
    const int voff = 2 * kNE + head * kHD;

    // ---- stage K (async, fragment order) ----
    {
        int s = tid, ntile = s >> 4, slr = s & 15;
        const unsigned short* kr = qk + (size_t)s * kQKV + koff;
#pragma unroll
        for (int c = 0; c < 8; ++c) {   // 8 halves per chunk along d
            int ks = c >> 2, cg = c & 1, j0 = ((c >> 1) & 1) * 4;
            async_copy16(&KF[((ks * 16 + ntile) * 32 + slr + 16 * cg) * 8 + j0], kr + c * 8);
        }
    }
    // ---- stage V (2-row packed dword scatter) ----
    {
        int p = tid & 127, dh = (tid >> 7) * 32;
        int s2 = 2 * p;
        const unsigned short* v0 = qk + (size_t)s2 * kQKV + voff + dh;
        const unsigned short* v1 = v0 + kQKV;
        alignas(16) unsigned short a0[32], a1[32];
#pragma unroll
        for (int c = 0; c < 4; ++c) {
            *(uint4*)&a0[c * 8] = *(const uint4*)(v0 + c * 8);
            *(uint4*)&a1[c * 8] = *(const uint4*)(v1 + c * 8);
        }
        int ks = p >> 4, dd = p & 15, vg = dd >> 3, vj = dd & 7;
#pragma unroll
        for (int e = 0; e < 32; ++e) {
            int d = dh + e;
            unsigned int dw = (unsigned int)a0[e] | ((unsigned int)a1[e] << 16);
            VF[((ks * 4 + (d >> 4)) * 32 + (d & 15) + 16 * vg) * 8 + vj] = dw;
        }
    }
    async_wait();
    __syncthreads();

    for (int qc = 0; qc < 4; ++qc) {
        const int qb = qc * 64;
        // ---- stage Q chunk (async, A-fragment order) ----
        {
            int row = tid >> 2, seg = tid & 3;
            int mtile = row >> 4, qlr = row & 15;
            const unsigned short* qr = qk + (size_t)(qb + row) * kQKV + qoff + seg * 16;
            int ks = seg >> 1, j0 = (seg & 1) * 4;
            async_copy16(&QF[((ks * 4 + mtile) * 32 + qlr) * 8 + j0], qr);
            async_copy16(&QF[((ks * 4 + mtile) * 32 + qlr + 16) * 8 + j0], qr + 8);
        }
        async_wait();
        __syncthreads();

        // ---- S = Q K^T / 8 ; wave = (16-row band) x (128-col half) ----
        {
            int band = wave & 3, nh = wave >> 2;
            int rbase = band * 16;
            Frag aq[2];
#pragma unroll
            for (int kk = 0; kk < 2; ++kk)
                load_frag8(aq[kk], &QF[((kk * 4 + band) * 32 + lane) * 8]);
#pragma unroll
            for (int nt = 0; nt < 8; ++nt) {
                int scol = nh * 128 + nt * 16;
                Frag bk0, bk1;
                load_frag8(bk0, &KF[((0 * 16 + nh * 8 + nt) * 32 + lane) * 8]);
                load_frag8(bk1, &KF[((1 * 16 + nh * 8 + nt) * 32 + lane) * 8]);
                v8f acc = vzero8();
                acc = bf16_wmma(aq[0], bk0, acc);
                acc = bf16_wmma(aq[1], bk1, acc);
#pragma unroll
                for (int r = 0; r < 8; ++r)
                    Sf[(rbase + r + 8 * grp) * 260 + scol + lr] = acc[r] * 0.125f;
            }
        }
        __syncthreads();

        // ---- softmax rows (deferred 1/sum normalization) ----
        {
#pragma unroll
            for (int rr = 0; rr < 8; ++rr) {
                int row = wave * 8 + rr;
                float* Sr = &Sf[row * 260];
                float vals[8], m = -1e30f;
#pragma unroll
                for (int i = 0; i < 8; ++i) {
                    vals[i] = Sr[lane + i * 32];
                    m = fmaxf(m, vals[i]);
                }
#pragma unroll
                for (int o = 16; o > 0; o >>= 1) m = fmaxf(m, __shfl_xor(m, o, 32));
                float ssum = 0.0f;
#pragma unroll
                for (int i = 0; i < 8; ++i) {
                    float e = __expf(vals[i] - m);
                    Sr[lane + i * 32] = e;
                    ssum += e;
                }
#pragma unroll
                for (int o = 16; o > 0; o >>= 1) ssum += __shfl_xor(ssum, o, 32);
                if (lane == 0) rsum[row] = 1.0f / ssum;
            }
        }
        __syncthreads();

        // ---- Y = P V ; wave = (16-row band) x (32 d-cols) ----
        {
            int band = wave & 3, dhw = wave >> 2;
            int rbase = band * 16;
            v8f acc[2] = {vzero8(), vzero8()};
            for (int ks = 0; ks < 8; ++ks) {
                int s0 = ks * 32;
                Frag ap;   // P: fp32 LDS -> packed bf16 pairs (A-fragment order)
#pragma unroll
                for (int j = 0; j < 8; ++j) {
                    int ka = 2 * j + 16 * grp + s0;
                    float2 f = *(const float2*)&Sf[(rbase + lr) * 260 + ka];
                    ap.u[j] = (unsigned int)f2bf(f.x) | ((unsigned int)f2bf(f.y) << 16);
                }
#pragma unroll
                for (int nt = 0; nt < 2; ++nt) {
                    Frag bv;
                    load_frag8(bv, &VF[((ks * 4 + dhw * 2 + nt) * 32 + lane) * 8]);
                    acc[nt] = bf16_wmma(ap, bv, acc[nt]);
                }
            }
#pragma unroll
            for (int nt = 0; nt < 2; ++nt) {
                int dcol = dhw * 32 + nt * 16 + lr;
#pragma unroll
                for (int r = 0; r < 8; ++r) {
                    int rowl = rbase + r + 8 * grp;
                    float val = acc[nt][r] * rsum[rowl];
                    size_t tok = (size_t)vb * kT + qb + rowl;
                    y[tok * kNE + head * kHD + dcol] = f2bf(val);
                }
            }
        }
        __syncthreads();
    }
}

// ---------------------------------------------------------------------------
// Host launcher
// ---------------------------------------------------------------------------
extern "C" void kernel_launch(void* const* d_in, const int* in_sizes, int n_in,
                              void* d_out, int out_size, void* d_ws, size_t ws_size,
                              hipStream_t stream) {
    (void)in_sizes; (void)n_in; (void)out_size; (void)ws_size;

    const float* x    = (const float*)d_in[0];
    const float* g1   = (const float*)d_in[1];
    const float* b1   = (const float*)d_in[2];
    const float* Wqkv = (const float*)d_in[3];
    const float* bqkv = (const float*)d_in[4];
    const float* Wp   = (const float*)d_in[5];
    const float* bp   = (const float*)d_in[6];
    const float* g2   = (const float*)d_in[7];
    const float* b2   = (const float*)d_in[8];
    const float* Wfc  = (const float*)d_in[9];
    const float* bfc  = (const float*)d_in[10];
    const float* Wm   = (const float*)d_in[11];
    const float* bm   = (const float*)d_in[12];
    float* out = (float*)d_out;

    char* ws = (char*)d_ws;
    float*          x1f   = (float*)(ws + OFF_X1F);
    unsigned short* x1h   = (unsigned short*)(ws + OFF_X1H);
    unsigned short* x2lnh = (unsigned short*)(ws + OFF_X1H);   // reuse (x1h dead)
    unsigned short* qkvh  = (unsigned short*)(ws + OFF_QKVH);
    unsigned short* yh    = (unsigned short*)(ws + OFF_YH);
    unsigned short* hh    = (unsigned short*)(ws + OFF_HH);    // reuse qkv+y region
    float*          x2f   = (float*)(ws + OFF_X2F);
    unsigned short* wqkvh = (unsigned short*)(ws + OFF_WQKV);
    unsigned short* wph   = (unsigned short*)(ws + OFF_WP);
    unsigned short* wfch  = (unsigned short*)(ws + OFF_WFC);
    unsigned short* wmh   = (unsigned short*)(ws + OFF_WM);

    auto cvt = [&](const float* s, unsigned short* d, int n) {
        cvt_bf16_kernel<<<(n + 255) / 256, 256, 0, stream>>>(s, d, n);
    };
    cvt(Wqkv, wqkvh, kNE * kQKV);
    cvt(Wp,   wph,   kNE * kNE);
    cvt(Wfc,  wfch,  kNE * kFF);
    cvt(Wm,   wmh,   kFF * kNE);

    // 1) transpose + LN1
    ln1_kernel<<<kTok, 256, 0, stream>>>(x, g1, b1, x1f, x1h);

    // 2) QKV GEMM
    gemm_bf16_kernel<0><<<dim3(kQKV / 128, kTok / 128), 256, 0, stream>>>(
        x1h, wqkvh, bqkv, nullptr, nullptr, qkvh, kTok, kQKV, kNE);

    // 3) attention
    (void)hipFuncSetAttribute(reinterpret_cast<const void*>(attn_kernel),
                              hipFuncAttributeMaxDynamicSharedMemorySize, ATTN_SMEM);
    attn_kernel<<<dim3(kHeads, kTok / kT), 256, ATTN_SMEM, stream>>>(qkvh, yh);

    // 4) attn proj + residual(x1) -> x2 fp32
    gemm_bf16_kernel<1><<<dim3(kNE / 128, kTok / 128), 256, 0, stream>>>(
        yh, wph, bp, x1f, x2f, nullptr, kTok, kNE, kNE);

    // 5) LN2 -> bf16
    ln2_kernel<<<kTok, 256, 0, stream>>>(x2f, g2, b2, x2lnh);

    // 6) FC + GELU -> bf16
    gemm_bf16_kernel<2><<<dim3(kFF / 128, kTok / 128), 256, 0, stream>>>(
        x2lnh, wfch, bfc, nullptr, nullptr, hh, kTok, kFF, kNE);

    // 7) MLP proj + residual(x2), scatter to [v,b,c,h,w]
    gemm_bf16_kernel<3><<<dim3(kNE / 128, kTok / 128), 256, 0, stream>>>(
        hh, wmh, bm, x2f, out, nullptr, kTok, kNE, kFF);
}
